// PhotonicImageEncoder_89069031784497
// MI455X (gfx1250) — matrix operand order repaired and checked by reading, not compiled, not run on hardware
//
#include <hip/hip_runtime.h>
#include <hip/hip_bf16.h>
#include <math.h>

typedef __attribute__((ext_vector_type(16))) _Float16 v16h;
typedef __attribute__((ext_vector_type(8)))  float    v8f;

#define PI_F 3.14159265358979323846f

// 7-way register select (lowered to v_cndmask chain; avoids scratch spills
// from runtime-indexed private arrays)
__device__ __forceinline__ float sel7(float v0, float v1, float v2, float v3,
                                      float v4, float v5, float v6, int i) {
  float r = v0;
  r = (i == 1) ? v1 : r;
  r = (i == 2) ? v2 : r;
  r = (i == 3) ? v3 : r;
  r = (i == 4) ? v4 : r;
  r = (i == 5) ? v5 : r;
  r = (i == 6) ? v6 : r;
  return r;
}
#define SEL7(a, i) sel7(a[0], a[1], a[2], a[3], a[4], a[5], a[6], (i))

__global__ __launch_bounds__(256)
void photonic_encode_kernel(const float* __restrict__ images,
                            unsigned long long* __restrict__ out) {
  // LDS: ~51 KB total (well under WGP budget)
  __shared__ float sFilt[32][52];    // gabor filters [f][k], k = i*7+j
  __shared__ float sPatch[128][52];  // 128 patches x 49 values (row stride 52)
  __shared__ float sGab[128][32];    // |gabor responses|
  __shared__ float sWfft[8][32];     // frequency-band weights (constants)
  __shared__ float sCt[8], sSt[8];   // 7-point DFT twiddles

  const int tid  = threadIdx.x;
  const int lane = tid & 31;
  const int wave = tid >> 5;
  const int PB   = blockIdx.x * 128;  // global patch base for this block
  const int LB   = wave * 16;         // local patch base for this wave

  // ---------------- Phase 0: constant tables ----------------
  if (tid < 32) {  // one filter per thread
    const int f = tid >> 3, o = tid & 7;
    const float freq = 2.0f + 2.0f * (float)f;
    const float th = (float)o * (PI_F / 8.0f);
    const float cth = __cosf(th), sth = __sinf(th);
    float mx = 0.0f;
#pragma unroll
    for (int k = 0; k < 49; ++k) {
      const int i = k / 7, j = k % 7;
      const float xx = (float)(i - 3) * (1.0f / 3.0f);
      const float yy = (float)(j - 3) * (1.0f / 3.0f);
      const float xr = xx * cth + yy * sth;  // rotation preserves xx^2+yy^2
      const float g = __expf(-(xx * xx + yy * yy) * (1.0f / 0.18f)) *
                      __cosf(2.0f * PI_F * freq * xr);
      sFilt[tid][k] = g;
      mx = fmaxf(mx, fabsf(g));
    }
    const float inv = 1.0f / (mx + 1e-8f);
#pragma unroll
    for (int k = 0; k < 49; ++k) sFilt[tid][k] *= inv;
  }
  {  // w_fft[c][m] = exp(-(m/31 - c/8)^2 / 0.1); 256 threads, one entry each
    const int c = tid >> 5, m = tid & 31;
    const float d = (float)m * (1.0f / 31.0f) - (float)c * 0.125f;
    sWfft[c][m] = __expf(-d * d * 10.0f);
  }
  if (tid < 7) {
    const float a = 2.0f * PI_F * (float)tid / 7.0f;
    sCt[tid] = __cosf(a);
    sSt[tid] = __sinf(a);
  }

  // ---------------- Phase A: stage 128 patches into LDS ----------------
  // images: [B,224,224]; patch p = by*32+bx of image b; element k = i*7+j
  for (int idx = tid; idx < 128 * 49; idx += 256) {
    const int lp = idx / 49;
    const int k  = idx - lp * 49;
    const int n  = PB + lp;
    const int b  = n >> 10;
    const int p  = n & 1023;
    const int by = p >> 5, bx = p & 31;
    const int row = by * 7 + k / 7;
    const int col = bx * 7 + (k % 7);
    sPatch[lp][k] = images[((size_t)b * 224 + row) * 224 + col];
  }
  __syncthreads();

  // ---------------- Phase B: gabor GEMM via WMMA ----------------
  // gab[16 patches x 32 filters] = A[16x49] * F^T[49x32], K padded to 64.
  {
    const int m    = lane & 15;
    const int kA   = (lane >> 4) * 8;   // A lane-group K offset
    const int n0   = lane & 15;
    const int kB   = (lane >> 4) * 16;  // B lane-group K offset
    v16h a0, a1, b00, b10, b01, b11;
#pragma unroll
    for (int v = 0; v < 8; ++v) {
      // A layout (16-bit 16x32): VGPR v<4 -> K = kA + 2v, v>=4 -> K = 16+kA+2(v-4)
      const int k0 = ((v >> 2) * 16) + kA + ((v & 3) * 2);
      const int k1 = k0 + 32;
      const int k0c = (k0 < 48) ? k0 : 48, k1c = (k1 < 48) ? k1 : 48;
      float f0 = sPatch[LB + m][k0c], f1 = sPatch[LB + m][k0c + 1];
      float g0 = sPatch[LB + m][k1c], g1 = sPatch[LB + m][k1c + 1];
      a0[2 * v]     = (_Float16)((k0     < 49) ? f0 : 0.0f);
      a0[2 * v + 1] = (_Float16)((k0 + 1 < 49) ? f1 : 0.0f);
      a1[2 * v]     = (_Float16)((k1     < 49) ? g0 : 0.0f);
      a1[2 * v + 1] = (_Float16)((k1 + 1 < 49) ? g1 : 0.0f);

      // B layout (32x16): same K order as A: VGPR v<4 -> K = 8h + 2v, v>=4 -> K = 16+8h+2(v-4)
      const int kb0 = ((v >> 2) * 16) + (kB >> 1) + ((v & 3) * 2);
      const int kb1 = kb0 + 32;
      const int kb0c = (kb0 < 48) ? kb0 : 48, kb1c = (kb1 < 48) ? kb1 : 48;
      float h00 = sFilt[n0][kb0c],      h01 = sFilt[n0][kb0c + 1];
      float h20 = sFilt[16 + n0][kb0c], h21 = sFilt[16 + n0][kb0c + 1];
      float h10 = sFilt[n0][kb1c],      h11 = sFilt[n0][kb1c + 1];
      float h30 = sFilt[16 + n0][kb1c], h31 = sFilt[16 + n0][kb1c + 1];
      b00[2 * v]     = (_Float16)((kb0     < 49) ? h00 : 0.0f);
      b00[2 * v + 1] = (_Float16)((kb0 + 1 < 49) ? h01 : 0.0f);
      b01[2 * v]     = (_Float16)((kb0     < 49) ? h20 : 0.0f);
      b01[2 * v + 1] = (_Float16)((kb0 + 1 < 49) ? h21 : 0.0f);
      b10[2 * v]     = (_Float16)((kb1     < 49) ? h10 : 0.0f);
      b10[2 * v + 1] = (_Float16)((kb1 + 1 < 49) ? h11 : 0.0f);
      b11[2 * v]     = (_Float16)((kb1     < 49) ? h30 : 0.0f);
      b11[2 * v + 1] = (_Float16)((kb1 + 1 < 49) ? h31 : 0.0f);
    }
    v8f acc0 = {}, acc1 = {};
    acc0 = __builtin_amdgcn_wmma_f32_16x16x32_f16(false, a0, false, b00, (short)0, acc0, false, false);
    acc0 = __builtin_amdgcn_wmma_f32_16x16x32_f16(false, a1, false, b10, (short)0, acc0, false, false);
    acc1 = __builtin_amdgcn_wmma_f32_16x16x32_f16(false, a0, false, b01, (short)0, acc1, false, false);
    acc1 = __builtin_amdgcn_wmma_f32_16x16x32_f16(false, a1, false, b11, (short)0, acc1, false, false);
    // C/D layout: VGPR r -> M = r (lanes 0-15) / 8+r (lanes 16-31), N = lane%16
#pragma unroll
    for (int r = 0; r < 8; ++r) {
      const int mrow = r + ((lane >> 4) * 8);
      sGab[LB + mrow][n0]      = fabsf(acc0[r]);
      sGab[LB + mrow][16 + n0] = fabsf(acc1[r]);
    }
  }
  __syncthreads();

  // ---------------- Phase C: per-patch channels, lane = mode ----------------
  const int m = lane;
  const float pos = (float)m * (6.0f / 31.0f);
  const int lo = (int)pos;
  const int hi = (lo + 1 < 7) ? lo + 1 : 6;
  const float fr = pos - (float)lo;
  const float ifr = 1.0f - fr;

  for (int pi = 0; pi < 16; ++pi) {
    const int lp = LB + pi;
    const size_t n = (size_t)(PB + lp);

    // One pass over the 7x7 patch: row0, column sums, blur row-sums, min/max
    float r0[7], cs[7], rs2[7], rs4[7], rs6[7];
    float mn = 1e30f, mx = -1e30f;
#pragma unroll
    for (int j = 0; j < 7; ++j) {
      const float p0 = sPatch[lp][j];
      const float p1 = sPatch[lp][7 + j];
      const float p2 = sPatch[lp][14 + j];
      const float p3 = sPatch[lp][21 + j];
      const float p4 = sPatch[lp][28 + j];
      const float p5 = sPatch[lp][35 + j];
      const float p6 = sPatch[lp][42 + j];
      r0[j]  = p0;
      cs[j]  = p0 + p1 + p2 + p3 + p4 + p5 + p6;
      rs2[j] = p0 + p1;                              // reflect rows {1,0}
      rs4[j] = p0 + 2.0f * p1 + p2;                  // rows {2,1,0,1}
      rs6[j] = p0 + 2.0f * p1 + 2.0f * p2 + p3;      // rows {3,2,1,0,1,2}
      float lmn = fminf(fminf(fminf(p0, p1), fminf(p2, p3)),
                        fminf(fminf(p4, p5), p6));
      float lmx = fmaxf(fmaxf(fmaxf(p0, p1), fmaxf(p2, p3)),
                        fmaxf(fmaxf(p4, p5), p6));
      mn = fminf(mn, lmn);
      mx = fmaxf(mx, lmx);
    }

    // 7-point DFT of column sums, only bins lo and hi needed
    float relo = 0.0f, imlo = 0.0f, rehi = 0.0f, imhi = 0.0f;
#pragma unroll
    for (int j = 0; j < 7; ++j) {
      const int il = (lo * j) % 7;
      const int ih = (hi * j) % 7;
      relo += cs[j] * sCt[il];  imlo -= cs[j] * sSt[il];
      rehi += cs[j] * sCt[ih];  imhi -= cs[j] * sSt[ih];
    }
    const float fftm = sqrtf(relo * relo + imlo * imlo) * ifr +
                       sqrtf(rehi * rehi + imhi * imhi) * fr;
    const float ph = atan2f(imlo, relo) * ifr + atan2f(imhi, rehi) * fr;

    const float pf = sPatch[lp][lo] * ifr + sPatch[lp][hi] * fr;

    const float inv = 1.0f / (mx - mn + 1e-8f);
    const float pnlo = (sPatch[lp][lo] - mn) * inv;
    const float pnhi = (sPatch[lp][hi] - mn) * inv;

    // Local-contrast row 0 at scales 1/2/4/6 (reflect padding, row 0 only)
    constexpr int m2c[9]  = {1, 0, 1, 2, 3, 4, 5, 6, 5};
    constexpr int m4c[11] = {2, 1, 0, 1, 2, 3, 4, 5, 6, 5, 4};
    constexpr int m6c[13] = {3, 2, 1, 0, 1, 2, 3, 4, 5, 6, 5, 4, 3};
    float c1x[7], c2x[7], c4x[7], c6x[7];
#pragma unroll
    for (int x = 0; x < 7; ++x) {
      const float b2 = 0.25f * (rs2[m2c[x]] + rs2[m2c[x + 1]]);
      float b4 = 0.0f;
#pragma unroll
      for (int d = 0; d < 4; ++d) b4 += rs4[m4c[x + d]];
      b4 *= (1.0f / 16.0f);
      float b6 = 0.0f;
#pragma unroll
      for (int d = 0; d < 6; ++d) b6 += rs6[m6c[x + d]];
      b6 *= (1.0f / 36.0f);
      c1x[x] = fabsf(r0[x]);
      c2x[x] = fabsf(r0[x] - b2);
      c4x[x] = fabsf(r0[x] - b4);
      c6x[x] = fabsf(r0[x] - b6);
    }
    const float ci1 = SEL7(c1x, lo) * ifr + SEL7(c1x, hi) * fr;
    const float ci2 = SEL7(c2x, lo) * ifr + SEL7(c2x, hi) * fr;
    const float ci4 = SEL7(c4x, lo) * ifr + SEL7(c4x, hi) * fr;
    const float ci6 = SEL7(c6x, lo) * ifr + SEL7(c6x, hi) * fr;

    // Streaming NT stores: one u64 = {re, im=0} per channel, lane = mode
    unsigned long long* obase = out + n * 2048ull + (unsigned)m;
#define STORE_CH(ch, val)                                                     \
  __builtin_nontemporal_store((unsigned long long)__float_as_uint(val),       \
                              obase + (size_t)(ch) * 32)

#pragma unroll
    for (int c = 0; c < 8; ++c) STORE_CH(c, sGab[lp][c] * pf);          // 0-7
#pragma unroll
    for (int c = 0; c < 8; ++c) STORE_CH(8 + c, fftm * sWfft[c][m]);    // 8-15
    {                                                                    // 16-23
      STORE_CH(16, ci1); STORE_CH(17, ci2); STORE_CH(18, ci4); STORE_CH(19, ci6);
      STORE_CH(20, ci6); STORE_CH(21, ci6); STORE_CH(22, ci6); STORE_CH(23, ci6);
    }
#pragma unroll
    for (int c = 0; c < 8; ++c) {                                        // 24-31
      const float cen = (float)c * (1.0f / 7.0f);
      const float dl = pnlo - cen, dh = pnhi - cen;
      const float val = __expf(-dl * dl * 12.5f) * ifr +
                        __expf(-dh * dh * 12.5f) * fr;
      STORE_CH(24 + c, val);
    }
#pragma unroll
    for (int c = 0; c < 16; ++c) STORE_CH(32 + c, sGab[lp][8 + c] * pf); // 32-47
#pragma unroll
    for (int c = 0; c < 16; ++c) {                                       // 48-63
      const float pb = (float)(c - 8) * (PI_F / 8.0f);
      const float d = ph - pb;
      STORE_CH(48 + c, __expf(-d * d * 2.0f) * pf);
    }
#undef STORE_CH
  }
}

extern "C" void kernel_launch(void* const* d_in, const int* in_sizes, int n_in,
                              void* d_out, int out_size, void* d_ws, size_t ws_size,
                              hipStream_t stream) {
  const float* images = (const float*)d_in[0];
  unsigned long long* out = (unsigned long long*)d_out;  // complex64 = u64 {re, im}
  const int B = in_sizes[0] / (224 * 224);               // 32
  // 1024 patches/image, 128 patches/block -> 8 blocks/image
  dim3 grid(B * 8);
  dim3 block(256);
  photonic_encode_kernel<<<grid, block, 0, stream>>>(images, out);
}
